// GAT_15135464751429
// MI455X (gfx1250) — matrix operand order repaired and checked by reading, not compiled, not run on hardware
//
#include <hip/hip_runtime.h>

#define H_  8
#define D_  16
#define HD_ 128

typedef __attribute__((ext_vector_type(16))) _Float16 v16h;
typedef __attribute__((ext_vector_type(8)))  _Float16 v8h;
typedef __attribute__((ext_vector_type(8)))  float    v8f;

__device__ __forceinline__ void atomicMaxFloat(float* addr, float v) {
  // valid for mixed-sign values when initialized to -inf (0xFF800000)
  if (v >= 0.0f) atomicMax((int*)addr, __float_as_int(v));
  else           atomicMin((unsigned int*)addr, __float_as_uint(v));
}

__global__ void k_cast_f16(const float* __restrict__ src, _Float16* __restrict__ dst, int count) {
  int i = blockIdx.x * blockDim.x + threadIdx.x;
  if (i < count) dst[i] = (_Float16)src[i];
}

// W is [128 x 128] row-major (K x N). Store transposed f16: Wt[n*128 + k] = W[k*128 + n]
// so each WMMA B-fragment lane (fixed column, contiguous K) is one aligned 32B read.
__global__ void k_cast_w_transpose(const float* __restrict__ W, _Float16* __restrict__ Wt) {
  int i = blockIdx.x * blockDim.x + threadIdx.x;   // over 128*128
  if (i >= HD_ * HD_) return;
  int k = i >> 7, c = i & (HD_ - 1);
  Wt[(size_t)c * HD_ + k] = (_Float16)W[i];
}

__global__ void k_zero_out(float* out) { out[0] = 0.0f; }

__global__ void k_init_accum(float* __restrict__ rst, float* __restrict__ m,
                             float* __restrict__ denom, int n) {
  int i = blockIdx.x * blockDim.x + threadIdx.x;
  if (i < n * HD_) rst[i] = 0.0f;
  if (i < n * H_) { m[i] = -__builtin_inff(); denom[i] = 0.0f; }
}

// C[n x 128] = A[n x 128](f16) * B(given transposed: Bt[n_col x 128 K])(f16), f32 acc.
// One block = 16-row strip; 8 waves, wave w owns 16-col tile w (== head w).
__global__ void __launch_bounds__(256) k_gemm_wmma(
    const _Float16* __restrict__ A, const _Float16* __restrict__ Bt,
    float* __restrict__ C, int n) {
  __shared__ __align__(16) _Float16 As[16 * 136];  // +8 halfs pad: LDS bank spread
  const int rowbase = blockIdx.x * 16;
  {
    const int t = threadIdx.x;
    const int r = t >> 4, part = t & 15;
    const int gr = rowbase + r;
    v8h val = {};
    if (gr < n) val = *(const v8h*)(A + (size_t)gr * HD_ + part * 8);
    *(v8h*)(As + r * 136 + part * 8) = val;
  }
  __syncthreads();

  const int wave    = threadIdx.x >> 5;
  const int lane    = threadIdx.x & 31;
  const int half_id = lane & 15;
  const bool hi     = lane >= 16;
  const int colbase = wave * 16;
  const int klo     = hi ? 16 : 0;   // A frag K sub-offset for upper half-wave
  const int kbase   = hi ? 16 : 0;   // B frag K base for upper half-wave

  // B column for this lane: 128 contiguous f16 in Bt
  const _Float16* bcol = Bt + (size_t)(colbase + half_id) * HD_;

  v8f acc = {};
#pragma unroll
  for (int kc = 0; kc < 4; ++kc) {
    const int k0 = kc * 32;
    // A fragment: row = half_id, K = k0+klo+{0..15}
    const _Float16* ap = As + half_id * 136 + k0 + klo;
    v8h alo = *(const v8h*)ap;
    v8h ahi = *(const v8h*)(ap + 8);
    v16h a;
#pragma unroll
    for (int i = 0; i < 8; ++i) { a[i] = alo[i]; a[i + 8] = ahi[i]; }
    // B fragment: col = colbase+half_id, K = k0+kbase+{0..15} -> contiguous 32B
    v16h b = *(const v16h*)(bcol + k0 + kbase);
    acc = __builtin_amdgcn_wmma_f32_16x16x32_f16(
        /*neg_a=*/false, a, /*neg_b=*/false, b,
        /*c_mod=*/(short)0, acc, /*reuse_a=*/false, /*reuse_b=*/false);
  }
  // C layout: VGPR k -> M = k (+8 for upper half-wave), N = half_id
  const int coln = colbase + half_id;
  float* cbase = C + (size_t)rowbase * HD_ + coln;
  const int moff = hi ? 8 : 0;
  if (rowbase + 16 <= n) {
    // Full strip (always taken when n % 16 == 0): branch-free store tail.
#pragma unroll
    for (int k = 0; k < 8; ++k) cbase[(size_t)(k + moff) * HD_] = acc[k];
  } else {
#pragma unroll
    for (int k = 0; k < 8; ++k) {
      const int mrow = k + moff;
      if (rowbase + mrow < n) cbase[(size_t)mrow * HD_] = acc[k];
    }
  }
}

__global__ void k_el_er(const float* __restrict__ feat, const float* __restrict__ al,
                        const float* __restrict__ ar, float* __restrict__ el,
                        float* __restrict__ er, int n) {
  int i = blockIdx.x * blockDim.x + threadIdx.x;
  if (i >= n * H_) return;
  int node = i >> 3, h = i & 7;
  const float* f = feat + (size_t)node * HD_ + h * D_;
  float sl = 0.f, sr = 0.f;
#pragma unroll
  for (int d = 0; d < D_; ++d) {
    float v = f[d];
    sl += v * al[h * D_ + d];
    sr += v * ar[h * D_ + d];
  }
  el[i] = sl; er[i] = sr;
}

__global__ void k_edge_max(const int* __restrict__ src, const int* __restrict__ dst,
                           const float* __restrict__ el, const float* __restrict__ er,
                           float* __restrict__ m, int e) {
  int i = blockIdx.x * blockDim.x + threadIdx.x;
  if (i >= e) return;
  int s = src[i], d = dst[i];
  const float* pl = el + (size_t)s * H_;
  const float* pr = er + (size_t)d * H_;
  float* pm = m + (size_t)d * H_;
#pragma unroll
  for (int h = 0; h < H_; ++h) {
    float v = pl[h] + pr[h];
    v = (v > 0.f) ? v : 0.2f * v;
    atomicMaxFloat(pm + h, v);
  }
}

__global__ void k_edge_denom(const int* __restrict__ src, const int* __restrict__ dst,
                             const float* __restrict__ el, const float* __restrict__ er,
                             const float* __restrict__ m, float* __restrict__ denom, int e) {
  int i = blockIdx.x * blockDim.x + threadIdx.x;
  if (i >= e) return;
  int s = src[i], d = dst[i];
  const float* pl = el + (size_t)s * H_;
  const float* pr = er + (size_t)d * H_;
  const float* pm = m + (size_t)d * H_;
  float* pd = denom + (size_t)d * H_;
#pragma unroll
  for (int h = 0; h < H_; ++h) {
    float v = pl[h] + pr[h];
    v = (v > 0.f) ? v : 0.2f * v;
    atomicAdd(pd + h, __expf(v - pm[h]));
  }
}

__global__ void k_edge_scatter(const int* __restrict__ src, const int* __restrict__ dst,
                               const float* __restrict__ el, const float* __restrict__ er,
                               const float* __restrict__ m, const float* __restrict__ den,
                               const float* __restrict__ feat, float* __restrict__ rst, int e) {
  int i = blockIdx.x * blockDim.x + threadIdx.x;
  if (i >= e * H_) return;
  int edge = i >> 3, h = i & 7;
  int s = src[edge], d = dst[edge];
  if (edge + 4096 < e) __builtin_prefetch(src + edge + 4096, 0, 0);
  float v = el[(size_t)s * H_ + h] + er[(size_t)d * H_ + h];
  v = (v > 0.f) ? v : 0.2f * v;
  float a = __expf(v - m[(size_t)d * H_ + h]) / den[(size_t)d * H_ + h];
  const float* fs = feat + (size_t)s * HD_ + h * D_;
  float* rd = rst + (size_t)d * HD_ + h * D_;
#pragma unroll
  for (int dd = 0; dd < D_; ++dd) atomicAdd(rd + dd, a * fs[dd]);
}

__global__ void k_finalize(const float* __restrict__ rst, const float* __restrict__ bias,
                           float* __restrict__ h32, _Float16* __restrict__ hh, int n) {
  int i = blockIdx.x * blockDim.x + threadIdx.x;
  if (i >= n * HD_) return;
  int f = i & (HD_ - 1);
  float v = rst[i] + bias[f];
  const float scale = 1.0507009873554805f;
  const float alpha = 1.6732632423543772f;
  v = (v > 0.f) ? scale * v : scale * alpha * (__expf(v) - 1.f);
  h32[i] = v;
  hh[i] = (_Float16)v;
}

__global__ void k_reduce(const float* __restrict__ h2, const float* __restrict__ mask,
                         float* __restrict__ out, int n) {
  __shared__ float sdata[256];
  int idx = blockIdx.x * 256 + threadIdx.x;
  float v = 0.f;
  if (idx < n) {
    const float4* p = (const float4*)(h2 + (size_t)idx * HD_);
    float s = 0.f;
#pragma unroll
    for (int i = 0; i < HD_ / 4; ++i) { float4 q = p[i]; s += q.x + q.y + q.z + q.w; }
    v = mask[idx] * s * (1.0f / (float)HD_);
  }
  sdata[threadIdx.x] = v;
  __syncthreads();
  for (int s2 = 128; s2 > 0; s2 >>= 1) {
    if (threadIdx.x < s2) sdata[threadIdx.x] += sdata[threadIdx.x + s2];
    __syncthreads();
  }
  if (threadIdx.x == 0) atomicAdd(out, sdata[0] / (float)n);
}

extern "C" void kernel_launch(void* const* d_in, const int* in_sizes, int n_in,
                              void* d_out, int out_size, void* d_ws, size_t ws_size,
                              hipStream_t stream) {
  const float* x    = (const float*)d_in[0];
  const float* mask = (const float*)d_in[1];
  const int*   src  = (const int*)d_in[2];
  const int*   dst  = (const int*)d_in[3];
  const float* W[2]  = {(const float*)d_in[4], (const float*)d_in[8]};
  const float* al[2] = {(const float*)d_in[5], (const float*)d_in[9]};
  const float* ar[2] = {(const float*)d_in[6], (const float*)d_in[10]};
  const float* bs[2] = {(const float*)d_in[7], (const float*)d_in[11]};
  float* out = (float*)d_out;

  const int n = in_sizes[0] / HD_;  // IN_FEATS == HD == 128
  const int e = in_sizes[2];

  char* ws = (char*)d_ws;
  size_t off = 0;
  auto take = [&](size_t bytes) -> void* {
    void* p = ws + off;
    off = (off + bytes + 255) & ~(size_t)255;
    return p;
  };
  float*    feat = (float*)take((size_t)n * HD_ * 4);
  float*    rst  = (float*)take((size_t)n * HD_ * 4);
  float*    h32  = (float*)take((size_t)n * HD_ * 4);
  _Float16* hh   = (_Float16*)take((size_t)n * HD_ * 2);
  float*    el   = (float*)take((size_t)n * H_ * 4);
  float*    er   = (float*)take((size_t)n * H_ * 4);
  float*    mbuf = (float*)take((size_t)n * H_ * 4);
  float*    den  = (float*)take((size_t)n * H_ * 4);
  _Float16* Wt[2];
  Wt[0] = (_Float16*)take((size_t)HD_ * HD_ * 2);
  Wt[1] = (_Float16*)take((size_t)HD_ * HD_ * 2);

  const int T = 256;
  k_cast_f16<<<(n * HD_ + T - 1) / T, T, 0, stream>>>(x, hh, n * HD_);
  k_cast_w_transpose<<<(HD_ * HD_ + T - 1) / T, T, 0, stream>>>(W[0], Wt[0]);
  k_cast_w_transpose<<<(HD_ * HD_ + T - 1) / T, T, 0, stream>>>(W[1], Wt[1]);
  k_zero_out<<<1, 1, 0, stream>>>(out);

  for (int L = 0; L < 2; ++L) {
    k_gemm_wmma<<<(n + 15) / 16, T, 0, stream>>>(hh, Wt[L], feat, n);
    k_el_er<<<(n * H_ + T - 1) / T, T, 0, stream>>>(feat, al[L], ar[L], el, er, n);
    k_init_accum<<<(n * HD_ + T - 1) / T, T, 0, stream>>>(rst, mbuf, den, n);
    k_edge_max<<<(e + T - 1) / T, T, 0, stream>>>(src, dst, el, er, mbuf, e);
    k_edge_denom<<<(e + T - 1) / T, T, 0, stream>>>(src, dst, el, er, mbuf, den, e);
    k_edge_scatter<<<(e * H_ + T - 1) / T, T, 0, stream>>>(src, dst, el, er, mbuf, den, feat, rst, e);
    k_finalize<<<(n * HD_ + T - 1) / T, T, 0, stream>>>(rst, bs[L], h32, hh, n);
  }
  k_reduce<<<(n + T - 1) / T, T, 0, stream>>>(h32, mask, out, n);
}
